// ScaledDotProductAttention_34643206210009
// MI455X (gfx1250) — matrix operand order repaired and checked by reading, not compiled, not run on hardware
//
#include <hip/hip_runtime.h>

// CDNA5 (gfx1250) wave32 WMMA types
typedef __attribute__((ext_vector_type(16))) _Float16 v16h;
typedef __attribute__((ext_vector_type(8)))  _Float16 v8h;
typedef __attribute__((ext_vector_type(8)))  float    v8f;

#define BB      2
#define HN      8
#define SS      4096
#define DD      64
#define WIN     512
#define QSTRIPE 128                    // q rows per 4-wave block
#define NQB     (SS / QSTRIPE)         // 32 stripes per (b,h)
#define KSTR    72                     // LDS K row stride in f16 (pad 64 -> 72, 16B aligned, conflict-free)
#define VSTR    40                     // LDS V^T row stride in f16 (pad 32 -> 40, 16B aligned, conflict-free)
// softmax scale folded with log2(e): (1/sqrt(64)) * 1.4426950408889634
#define SCALE_LOG2E 0.18033688011112043f
#define MASKVAL (-1.0e30f)

// Stage one 32-row K tile (row-major f16) and V tile (transposed f16) into LDS.
// 128 threads: thread t handles source row t>>2, d-segment (t&3)*16.
__device__ __forceinline__ void stage_kv(
    int tid, int kb,
    const float* __restrict__ Kp, const float* __restrict__ Vp,
    _Float16* __restrict__ Kf, _Float16* __restrict__ VT)
{
    const int kr = tid >> 2;           // 0..31
    const int db = (tid & 3) * 16;     // 0,16,32,48

    // ---- K: row-major f16 ----
    const float* ks = Kp + (size_t)(kb + kr) * DD + db;
    const float4 a = *(const float4*)(ks + 0);
    const float4 b = *(const float4*)(ks + 4);
    const float4 c = *(const float4*)(ks + 8);
    const float4 d = *(const float4*)(ks + 12);
    v8h lo, hi;
    lo[0] = (_Float16)a.x; lo[1] = (_Float16)a.y; lo[2] = (_Float16)a.z; lo[3] = (_Float16)a.w;
    lo[4] = (_Float16)b.x; lo[5] = (_Float16)b.y; lo[6] = (_Float16)b.z; lo[7] = (_Float16)b.w;
    hi[0] = (_Float16)c.x; hi[1] = (_Float16)c.y; hi[2] = (_Float16)c.z; hi[3] = (_Float16)c.w;
    hi[4] = (_Float16)d.x; hi[5] = (_Float16)d.y; hi[6] = (_Float16)d.z; hi[7] = (_Float16)d.w;
    *(v8h*)(Kf + kr * KSTR + db)     = lo;
    *(v8h*)(Kf + kr * KSTR + db + 8) = hi;

    // ---- V: transposed f16, VT[d][krow] ----
    const float* vs = Vp + (size_t)(kb + kr) * DD + db;
    const float4 e  = *(const float4*)(vs + 0);
    const float4 f  = *(const float4*)(vs + 4);
    const float4 g  = *(const float4*)(vs + 8);
    const float4 h2 = *(const float4*)(vs + 12);
    float vv[16];
    vv[0]  = e.x;  vv[1]  = e.y;  vv[2]  = e.z;  vv[3]  = e.w;
    vv[4]  = f.x;  vv[5]  = f.y;  vv[6]  = f.z;  vv[7]  = f.w;
    vv[8]  = g.x;  vv[9]  = g.y;  vv[10] = g.z;  vv[11] = g.w;
    vv[12] = h2.x; vv[13] = h2.y; vv[14] = h2.z; vv[15] = h2.w;
    #pragma unroll
    for (int i = 0; i < 16; ++i)
        VT[(db + i) * VSTR + kr] = (_Float16)vv[i];
}

// One 32-column k-step for a wave owning 32 q rows (two 16-row sub-tiles).
// MODE: 0 = interior (no mask), 1 = left window edge, 2 = causal edge.
template<int MODE>
__device__ __forceinline__ void attn_step(
    int kb, int q0, int ln, int hh,
    const _Float16* __restrict__ Kf, const _Float16* __restrict__ VT,
    const v16h (&bq)[2][2], v8f (&oacc)[2][4],
    float (&m)[2], float (&l)[2])
{
    // ---- K A-tiles from LDS (A layout: lane M = ln; elem e -> K=16hh+e)
    v16h ak[2][2];
    #pragma unroll
    for (int t = 0; t < 2; ++t) {
        const _Float16* kr = Kf + (16 * t + ln) * KSTR;
        #pragma unroll
        for (int dh = 0; dh < 2; ++dh) {
            const v8h a0 = *(const v8h*)(kr + dh * 32 + hh * 16);
            const v8h a1 = *(const v8h*)(kr + dh * 32 + hh * 16 + 8);
            ak[t][dh] = __builtin_shufflevector(a0, a1,
                            0,1,2,3,4,5,6,7,8,9,10,11,12,13,14,15);
        }
    }

    // ---- per sub-tile: S^T = K x Q^T, online softmax, pack P directly as A --
    v16h ap[2];
    #pragma unroll
    for (int u = 0; u < 2; ++u) {
        const int iq = q0 + 16 * u + ln;           // this lane's q row
        v8f st[2];
        #pragma unroll
        for (int t = 0; t < 2; ++t) {
            #pragma unroll
            for (int r = 0; r < 8; ++r) st[t][r] = 0.0f;
            st[t] = __builtin_amdgcn_wmma_f32_16x16x32_f16(
                        false, ak[t][0], false, bq[u][0], (short)0, st[t], false, false);
            st[t] = __builtin_amdgcn_wmma_f32_16x16x32_f16(
                        false, ak[t][1], false, bq[u][1], (short)0, st[t], false, false);
        }
        // C layout: lane holds q col = iq; rows = kcol_local = 16t + 8hh + r
        float p[2][8];
        float smax = MASKVAL;
        #pragma unroll
        for (int t = 0; t < 2; ++t)
            #pragma unroll
            for (int r = 0; r < 8; ++r) {
                float s = st[t][r] * SCALE_LOG2E;
                if (MODE == 1) {                   // only the window bound can cut
                    const int j = kb + 16 * t + 8 * hh + r;
                    s = (j >= iq - (WIN - 1)) ? s : MASKVAL;
                } else if (MODE == 2) {            // only causality can cut
                    const int j = kb + 16 * t + 8 * hh + r;
                    s = (j <= iq) ? s : MASKVAL;
                }
                p[t][r] = s;
                smax = fmaxf(smax, s);
            }
        smax = fmaxf(smax, __shfl_xor(smax, 16, 32));
        const float mnew  = fmaxf(m[u], smax);
        const float alpha = exp2f(m[u] - mnew);
        float rsum = 0.0f;
        v16h a;
        #pragma unroll
        for (int t = 0; t < 2; ++t)
            #pragma unroll
            for (int r = 0; r < 8; ++r) {
                const float e = exp2f(p[t][r] - mnew);
                rsum += e;
                // C layout of S^T == A layout of P for O = P*V: a[t*8+r]
                a[t * 8 + r] = (_Float16)e;
            }
        ap[u] = a;
        rsum += __shfl_xor(rsum, 16, 32);
        l[u] = l[u] * alpha + rsum;
        m[u] = mnew;
        #pragma unroll
        for (int c = 0; c < 4; ++c)
            #pragma unroll
            for (int r = 0; r < 8; ++r) oacc[u][c][r] *= alpha;
    }

    // ---- O += P x V : B operand straight out of staged V^T (shared by both u)
    #pragma unroll
    for (int c = 0; c < 4; ++c) {
        // B layout: lane N = d = c*16 + ln; elem e<8 -> krow = 8hh + e, e>=8 -> krow = 16 + 8hh + e-8
        const _Float16* vr = VT + (c * 16 + ln) * VSTR + 8 * hh;
        const v8h b0 = *(const v8h*)(vr);
        const v8h b1 = *(const v8h*)(vr + 16);
        const v16h bv = __builtin_shufflevector(b0, b1,
                            0,1,2,3,4,5,6,7,8,9,10,11,12,13,14,15);
        oacc[0][c] = __builtin_amdgcn_wmma_f32_16x16x32_f16(
                         false, ap[0], false, bv, (short)0, oacc[0][c], false, false);
        oacc[1][c] = __builtin_amdgcn_wmma_f32_16x16x32_f16(
                         false, ap[1], false, bv, (short)0, oacc[1][c], false, false);
    }
}

__global__ __launch_bounds__(128, 1)
void swa_attn_fwd_f16wmma(const float* __restrict__ Q,
                          const float* __restrict__ K,
                          const float* __restrict__ V,
                          float* __restrict__ O)
{
    const int tid  = threadIdx.x;
    const int lane = tid & 31;
    // Wave index is uniform across the wave: pin it to an SGPR so the per-step
    // mode dispatch compiles to scalar branches (EXEC untouched around WMMA).
    const int wv   = __builtin_amdgcn_readfirstlane(tid >> 5);
    const int bh   = blockIdx.x / NQB;            // fused (b,h)
    const int Q0   = (blockIdx.x % NQB) * QSTRIPE;
    const int q0   = Q0 + wv * 32;                // this wave's 32 q rows (SGPR)

    const size_t base = (size_t)bh * SS * DD;
    const float* Qp = Q + base;
    const float* Kp = K + base;
    const float* Vp = V + base;
    float*       Op = O + base;

    const int ln = lane & 15;
    const int hh = lane >> 4;

    __shared__ __align__(16) _Float16 Kf[2][32 * KSTR];
    __shared__ __align__(16) _Float16 VT[2][64 * VSTR];

    // ---- Preload Q^T as B operands: 2 sub-tiles x 2 d-halves ---------------
    // B layout: lane N = ln; element e of v16h = K 16*hh + e
    v16h bq[2][2];
    #pragma unroll
    for (int u = 0; u < 2; ++u) {
        const float* qrow = Qp + (size_t)(q0 + 16 * u + ln) * DD;
        #pragma unroll
        for (int dh = 0; dh < 2; ++dh) {
            const float* src = qrow + dh * 32 + hh * 16;
            const float4 f0 = *(const float4*)(src + 0);
            const float4 f1 = *(const float4*)(src + 4);
            const float4 f2 = *(const float4*)(src + 8);
            const float4 f3 = *(const float4*)(src + 12);
            v16h b;
            b[0]  = (_Float16)f0.x; b[1]  = (_Float16)f0.y;
            b[2]  = (_Float16)f0.z; b[3]  = (_Float16)f0.w;
            b[4]  = (_Float16)f1.x; b[5]  = (_Float16)f1.y;
            b[6]  = (_Float16)f1.z; b[7]  = (_Float16)f1.w;
            b[8]  = (_Float16)f2.x; b[9]  = (_Float16)f2.y;
            b[10] = (_Float16)f2.z; b[11] = (_Float16)f2.w;
            b[12] = (_Float16)f3.x; b[13] = (_Float16)f3.y;
            b[14] = (_Float16)f3.z; b[15] = (_Float16)f3.w;
            bq[u][dh] = b;
        }
    }

    v8f oacc[2][4];
    #pragma unroll
    for (int u = 0; u < 2; ++u)
        #pragma unroll
        for (int c = 0; c < 4; ++c)
            #pragma unroll
            for (int r = 0; r < 8; ++r) oacc[u][c][r] = 0.0f;
    float m[2] = {MASKVAL, MASKVAL};
    float l[2] = {0.0f, 0.0f};

    // ---- shared k loop over the 4-wave union window, double-buffered --------
    int ks = Q0 - WIN; if (ks < 0) ks = 0;
    const int ke = Q0 + QSTRIPE;

    stage_kv(tid, ks, Kp, Vp, Kf[0], VT[0]);
    int buf = 0;
    for (int kb = ks; kb < ke; kb += 32) {
        __syncthreads();                            // buf staged; buf^1 free
        if (kb + 32 < ke) {
            if (kb + 64 < ke) {
                __builtin_prefetch(Kp + (size_t)(kb + 64 + (tid >> 2)) * DD + (tid & 3) * 16, 0, 1);
                __builtin_prefetch(Vp + (size_t)(kb + 64 + (tid >> 2)) * DD + (tid & 3) * 16, 0, 1);
            }
            stage_kv(tid, kb + 32, Kp, Vp, Kf[buf ^ 1], VT[buf ^ 1]);
        }
        // scalar (SGPR) mode dispatch; inactive waves only stage + barrier
        if (kb >= q0 - WIN && kb <= q0) {
            if (kb == q0)
                attn_step<2>(kb, q0, ln, hh, Kf[buf], VT[buf], bq, oacc, m, l);
            else if (kb == q0 - WIN)
                attn_step<1>(kb, q0, ln, hh, Kf[buf], VT[buf], bq, oacc, m, l);
            else
                attn_step<0>(kb, q0, ln, hh, Kf[buf], VT[buf], bq, oacc, m, l);
        }
        buf ^= 1;
    }

    // ---- epilogue: O[q][d] = O / l ; l fetched per output row via bpermute --
    #pragma unroll
    for (int u = 0; u < 2; ++u) {
        float rl[8];
        #pragma unroll
        for (int r = 0; r < 8; ++r)
            rl[r] = 1.0f / __shfl(l[u], r + 8 * hh, 32);
        float* ob = Op + (size_t)(q0 + 16 * u + 8 * hh) * DD;
        #pragma unroll
        for (int c = 0; c < 4; ++c)
            #pragma unroll
            for (int r = 0; r < 8; ++r)
                ob[(size_t)r * DD + c * 16 + ln] = oacc[u][c][r] * rl[r];
    }
}

extern "C" void kernel_launch(void* const* d_in, const int* in_sizes, int n_in,
                              void* d_out, int out_size, void* d_ws, size_t ws_size,
                              hipStream_t stream) {
    (void)in_sizes; (void)n_in; (void)out_size; (void)d_ws; (void)ws_size;
    const float* Q = (const float*)d_in[0];
    const float* K = (const float*)d_in[1];
    const float* V = (const float*)d_in[2];
    float* O = (float*)d_out;
    // 16 (b,h) x 32 stripes of 128 q rows; 128 threads (4 waves) per block
    const int blocks = BB * HN * NQB;   // 512
    swa_attn_fwd_f16wmma<<<blocks, 128, 0, stream>>>(Q, K, V, O);
}
